// GraphAttentionLayer_11123965297101
// MI455X (gfx1250) — matrix operand order repaired and checked by reading, not compiled, not run on hardware
//
#include <hip/hip_runtime.h>
#include <math.h>

// GAT fused kernel for MI455X (gfx1250, wave32, WMMA).
// Score matrices are rank-1 (s_i + t_j) before lrelu/mask, so the heavy work is
// 4x fused masked-softmax attention GEMMs (8192x8192x64) on v_wmma_f32_16x16x32_f16,
// bounded by ~1GB of streaming mask reads (~43us @ 23.3 TB/s HBM).
// All four branches run in one kernel (blockIdx.y) for 4x wave occupancy; mask
// reads use non-temporal hints so the 192MB L2 keeps the hot Bpack/t vectors.

#define NN   8192
#define KK   8192
#define FF   64

typedef __attribute__((ext_vector_type(16))) _Float16 v16h;
typedef __attribute__((ext_vector_type(8)))  float    v8f;
typedef __attribute__((ext_vector_type(4)))  float    v4f;
typedef __attribute__((ext_vector_type(4)))  int      v4i;

// ---------------------------------------------------------------------------
// Wh = A @ W  (A: [8192,K] f32 row-major, W: [K,64] f32)  — exact f32
__global__ __launch_bounds__(256)
void gat_gemm_f32(const float* __restrict__ A, const float* __restrict__ W,
                  float* __restrict__ C, int K)
{
    const int t = threadIdx.x;
    const int f = t & 63;
    const int r = blockIdx.x * 4 + (t >> 6);
    float acc = 0.f;
    for (int k = 0; k < K; ++k)
        acc = fmaf(A[(size_t)r * K + k], W[k * 64 + f], acc);
    C[(size_t)r * 64 + f] = acc;
}

// ---------------------------------------------------------------------------
// Per-row score scalars for all 4 branches (s = Wh@a[:64], t = Wh@a[64:])
__global__ __launch_bounds__(256)
void gat_score_vecs(const float* __restrict__ Whn, const float* __restrict__ Whe,
                    const float* __restrict__ a_n, const float* __restrict__ a_e,
                    const float* __restrict__ q_e, const float* __restrict__ q_n,
                    float* __restrict__ vecs)
{
    const int i = blockIdx.x * blockDim.x + threadIdx.x;
    const float* hn = Whn + (size_t)i * 64;
    const float* he = Whe + (size_t)i * 64;
    float s_nn = 0, t_nn = 0, s_en = 0, t_en = 0;
    float s_ee = 0, t_ee = 0, s_ne = 0, t_ne = 0;
    for (int k = 0; k < 64; ++k) {
        const float xn = hn[k], xe = he[k];
        s_nn = fmaf(xn, a_n[k],      s_nn);  t_nn = fmaf(xn, a_n[64 + k], t_nn);
        s_en = fmaf(xn, a_e[k],      s_en);  t_en = fmaf(xe, a_e[64 + k], t_en);
        s_ee = fmaf(xe, q_e[k],      s_ee);  t_ee = fmaf(xe, q_e[64 + k], t_ee);
        s_ne = fmaf(xe, q_n[k],      s_ne);  t_ne = fmaf(xn, q_n[64 + k], t_ne);
    }
    vecs[0 * NN + i] = s_nn;  vecs[1 * NN + i] = t_nn;
    vecs[2 * NN + i] = s_en;  vecs[3 * NN + i] = t_en;
    vecs[4 * NN + i] = s_ee;  vecs[5 * NN + i] = t_ee;
    vecs[6 * NN + i] = s_ne;  vecs[7 * NN + i] = t_ne;
}

// ---------------------------------------------------------------------------
// Global max of each t-vector (blockIdx.x = branch 0..3 -> vecs[2b+1])
__global__ __launch_bounds__(256)
void gat_tmax(const float* __restrict__ vecs, float* __restrict__ maxT)
{
    __shared__ float sm[256];
    const float* tv = vecs + (size_t)(2 * blockIdx.x + 1) * NN;
    float mx = -3.4e38f;
    for (int k = threadIdx.x; k < NN; k += 256) mx = fmaxf(mx, tv[k]);
    sm[threadIdx.x] = mx;
    __syncthreads();
    for (int off = 128; off > 0; off >>= 1) {
        if (threadIdx.x < off) sm[threadIdx.x] = fmaxf(sm[threadIdx.x], sm[threadIdx.x + off]);
        __syncthreads();
    }
    if (threadIdx.x == 0) maxT[blockIdx.x] = sm[0];
}

// ---------------------------------------------------------------------------
// Pre-pack Wh (f32 [8192,64]) into the ISA WMMA B-fragment layout, f16.
// idx = ((kb*4 + t)*32 + lane); slot holds 16 halves: K = kb*32 + 8*(lane/16) + h + (h>=8 ? 8 : 0),
// col = t*16 + lane%16.  Main loop then does pure 32B vector loads.
__global__ __launch_bounds__(256)
void gat_pack_b(const float* __restrict__ Wh, _Float16* __restrict__ Bp)
{
    const int idx  = blockIdx.x * blockDim.x + threadIdx.x;   // 32768
    const int lane = idx & 31;
    const int tt   = (idx >> 5) & 3;
    const int kb   = idx >> 7;
    const int col  = tt * 16 + (lane & 15);
    const int k0   = kb * 32 + 8 * (lane >> 4);
    _Float16* dst  = Bp + (size_t)idx * 16;
#pragma unroll
    for (int h = 0; h < 16; ++h)
        dst[h] = (_Float16)Wh[(size_t)(k0 + h + ((h >= 8) ? 8 : 0)) * 64 + col];
}

// ---------------------------------------------------------------------------
struct BranchArgs {
    const float*    sV;
    const float*    tV;
    const float*    maxTp;
    const int*      mask;
    const _Float16* Bp;
    float*          outp;   // pre-offset to branch column base
    int             trans;  // mask element (i,k) read as mask[k*NN + i]
};
struct AllArgs { BranchArgs b[4]; };

// Fused masked-softmax attention GEMM body:
//   out[i,:] = softmax_j(mask, lrelu(s_i + t_j)) @ Wh
// One wave = one 16-row x 64-col output tile; K-steps of 32 with 4 WMMAs each.
template <bool TRANS>
__device__ __forceinline__ void gat_attn_body(const BranchArgs ba)
{
    const int lane = threadIdx.x & 31;
    const int wave = threadIdx.x >> 5;
    const int i0   = (blockIdx.x * 8 + wave) * 16;
    const int m    = lane & 15;
    const int hi   = lane >> 4;
    const int i    = i0 + m;
    const int koB  = 8 * hi;           // lane-half K offset (ISA A-layout)

    const float s  = ba.sV[i];
    const float mT = ba.maxTp[0];
    const float mR = fmaxf(s + mT, 0.2f * (s + mT)); // lrelu(s+maxT) >= all row scores

    v8f acc0 = {}, acc1 = {}, acc2 = {}, acc3 = {};
    float psum = 0.f;

    for (int j = 0; j < KK; j += 32) {
        // t values for this lane's 16 K slots (reused by every wave -> cached)
        float tv[16];
        {
            const v4f ta = *(const v4f*)(ba.tV + j + koB);
            const v4f tb = *(const v4f*)(ba.tV + j + koB + 4);
            const v4f tc = *(const v4f*)(ba.tV + j + 16 + koB);
            const v4f td = *(const v4f*)(ba.tV + j + 16 + koB + 4);
#pragma unroll
            for (int q = 0; q < 4; ++q) {
                tv[q] = ta[q]; tv[4 + q] = tb[q]; tv[8 + q] = tc[q]; tv[12 + q] = td[q];
            }
        }
        int mk[16];
        if (!TRANS) {
            // streaming, consumed-once: non-temporal so L2 keeps hot data
            const int* mrow = ba.mask + (size_t)i * KK;
            const v4i ma = __builtin_nontemporal_load((const v4i*)(mrow + j + koB));
            const v4i mb = __builtin_nontemporal_load((const v4i*)(mrow + j + koB + 4));
            const v4i mc = __builtin_nontemporal_load((const v4i*)(mrow + j + 16 + koB));
            const v4i md = __builtin_nontemporal_load((const v4i*)(mrow + j + 16 + koB + 4));
#pragma unroll
            for (int q = 0; q < 4; ++q) {
                mk[q] = ma[q]; mk[4 + q] = mb[q]; mk[8 + q] = mc[q]; mk[12 + q] = md[q];
            }
        } else {
            // transposed mask: lines are shared with the neighboring row-tile -> keep cached
#pragma unroll
            for (int h = 0; h < 16; ++h) {
                const int k = j + ((h < 8) ? (koB + h) : (16 + koB + (h - 8)));
                mk[h] = ba.mask[(size_t)k * NN + i];
            }
        }
        v16h afrag;
#pragma unroll
        for (int h = 0; h < 16; ++h) {
            const float x = s + tv[h];
            const float e = fmaxf(x, 0.2f * x);                 // leaky relu
            const float w = (mk[h] > 0) ? __expf(e - mR) : 0.f; // masked exp, <= 1
            const _Float16 wh = (_Float16)w;
            psum += (float)wh;                                  // sum matches f16 numerator
            afrag[h] = wh;
        }
        const _Float16* bb = ba.Bp + (size_t)(j >> 5) * (4 * 32 * 16) + lane * 16;
        const v16h b0 = *(const v16h*)(bb + 0 * 32 * 16);
        const v16h b1 = *(const v16h*)(bb + 1 * 32 * 16);
        const v16h b2 = *(const v16h*)(bb + 2 * 32 * 16);
        const v16h b3 = *(const v16h*)(bb + 3 * 32 * 16);
        acc0 = __builtin_amdgcn_wmma_f32_16x16x32_f16(false, afrag, false, b0, (short)0, acc0, false, false);
        acc1 = __builtin_amdgcn_wmma_f32_16x16x32_f16(false, afrag, false, b1, (short)0, acc1, false, false);
        acc2 = __builtin_amdgcn_wmma_f32_16x16x32_f16(false, afrag, false, b2, (short)0, acc2, false, false);
        acc3 = __builtin_amdgcn_wmma_f32_16x16x32_f16(false, afrag, false, b3, (short)0, acc3, false, false);
    }

    // row sum: lanes L and L^16 hold complementary K halves of row L%16
    const float total = psum + __shfl_xor(psum, 16, 32);
    const float rinv  = 1.0f / total;

    // D layout: VGPR r of lane L -> (row i0 + r + 8*(L/16), col tile*16 + L%16)
#pragma unroll
    for (int r = 0; r < 8; ++r) {
        const float ri  = __shfl(rinv, r + 8 * hi, 32);
        const int   row = i0 + r + 8 * hi;
        float v[4] = { acc0[r] * ri, acc1[r] * ri, acc2[r] * ri, acc3[r] * ri };
#pragma unroll
        for (int tt = 0; tt < 4; ++tt) {
            float x = v[tt];
            x = (x > 0.f) ? x : (__expf(x) - 1.f);              // ELU
            __builtin_nontemporal_store(x, ba.outp + (size_t)row * 128 + tt * 16 + m);
        }
    }
}

// All four branches in one launch (blockIdx.y = branch) -> 2048 live waves.
// The trans/non-trans selector is a uniform scalar branch: EXEC stays all-ones
// around the WMMAs as the ISA requires.
__global__ __launch_bounds__(256)
void gat_attn_fused_all(AllArgs args)
{
    const BranchArgs ba = args.b[blockIdx.y];
    if (ba.trans) gat_attn_body<true>(ba);
    else          gat_attn_body<false>(ba);
}

// ---------------------------------------------------------------------------
extern "C" void kernel_launch(void* const* d_in, const int* in_sizes, int n_in,
                              void* d_out, int out_size, void* d_ws, size_t ws_size,
                              hipStream_t stream) {
    (void)in_sizes; (void)n_in; (void)out_size; (void)ws_size;

    const float* h        = (const float*)d_in[0];
    const float* edge_h   = (const float*)d_in[1];
    const int*   adj      = (const int*)d_in[2];
    const int*   edge_adj = (const int*)d_in[3];
    const int*   ne_adj   = (const int*)d_in[4];
    const float* W_n      = (const float*)d_in[5];
    const float* a_n      = (const float*)d_in[6];
    const float* W_e      = (const float*)d_in[7];
    const float* a_e      = (const float*)d_in[8];
    const float* q_e      = (const float*)d_in[9];
    const float* q_n      = (const float*)d_in[10];
    float* out = (float*)d_out;

    char* ws = (char*)d_ws;
    float*    Whn  = (float*)ws;     ws += (size_t)NN * 64 * 4;   // 2 MB
    float*    Whe  = (float*)ws;     ws += (size_t)NN * 64 * 4;   // 2 MB
    _Float16* Bpn  = (_Float16*)ws;  ws += (size_t)NN * 64 * 2;   // 1 MB
    _Float16* Bpe  = (_Float16*)ws;  ws += (size_t)NN * 64 * 2;   // 1 MB
    float*    vecs = (float*)ws;     ws += (size_t)8 * NN * 4;    // 256 KB
    float*    maxT = (float*)ws;     ws += 256;

    gat_gemm_f32<<<NN / 4, 256, 0, stream>>>(h,      W_n, Whn, 256);
    gat_gemm_f32<<<NN / 4, 256, 0, stream>>>(edge_h, W_e, Whe, 64);
    gat_score_vecs<<<NN / 256, 256, 0, stream>>>(Whn, Whe, a_n, a_e, q_e, q_n, vecs);
    gat_tmax<<<4, 256, 0, stream>>>(vecs, maxT);
    gat_pack_b<<<128, 256, 0, stream>>>(Whn, Bpn);
    gat_pack_b<<<128, 256, 0, stream>>>(Whe, Bpe);

    const size_t eBase = (size_t)NN * 128;
    AllArgs aa;
    // n_n: att(adj) @ Wh_n             -> out_n[:, 0:64]
    aa.b[0] = { vecs + 0 * NN, vecs + 1 * NN, maxT + 0, adj,      Bpn, out + 0,           0 };
    // e_n: att(node_edge_adj.T) @ Wh_e -> out_n[:, 64:128]
    aa.b[1] = { vecs + 2 * NN, vecs + 3 * NN, maxT + 1, ne_adj,   Bpe, out + 64,          1 };
    // e_e: att(edge_adj) @ Wh_e        -> out_e[:, 64:128]
    aa.b[2] = { vecs + 4 * NN, vecs + 5 * NN, maxT + 2, edge_adj, Bpe, out + eBase + 64,  0 };
    // n_e: att(node_edge_adj) @ Wh_n   -> out_e[:, 0:64]
    aa.b[3] = { vecs + 6 * NN, vecs + 7 * NN, maxT + 3, ne_adj,   Bpn, out + eBase + 0,   0 };

    // 512 row-tiles per branch, 8 waves/block -> grid (64, 4)
    gat_attn_fused_all<<<dim3(64, 4), 256, 0, stream>>>(aa);
}
